// MultiHeadAttention_6236292513970
// MI455X (gfx1250) — matrix operand order repaired and checked by reading, not compiled, not run on hardware
//
#include <hip/hip_runtime.h>
#include <math.h>

typedef __attribute__((ext_vector_type(16))) _Float16 v16h;
typedef __attribute__((ext_vector_type(8)))  _Float16 v8h;
typedef __attribute__((ext_vector_type(4)))  _Float16 v4h;
typedef __attribute__((ext_vector_type(8)))  float    v8f;

#define D_MODEL   1024
#define NUM_HEADS 16
#define D_HEAD    64
#define BATCH     2
#define SEQ       2048
#define BS        (BATCH * SEQ)   /* 4096 */

#define QLDS_OFF  201856          /* LDS offset of staged Q tile */
#define LDS_BYTES 203904

static __device__ __forceinline__ v8f wmma16x16x32(v16h a, v16h b, v8f c) {
    return __builtin_amdgcn_wmma_f32_16x16x32_f16(
        /*neg_a=*/false, a, /*neg_b=*/false, b,
        /*c_mod=*/(short)0, c, /*reuse_a=*/false, /*reuse_b=*/false);
}

static __device__ __forceinline__ v16h combine(v8h x, v8h y) {
    v16h r;
#pragma unroll
    for (int e = 0; e < 8; ++e) { r[e] = x[e]; r[8 + e] = y[e]; }
    return r;
}

// ---------------------------------------------------------------------------
// Streaming f32 -> f16 conversion (coalesced: 32B load, 16B store per thread)
// ---------------------------------------------------------------------------
__global__ void cvt_kernel(const float* __restrict__ in, _Float16* __restrict__ out,
                           long n8) {
    const long i = (long)blockIdx.x * blockDim.x + threadIdx.x;
    if (i >= n8) return;
    const long base = i * 8;
    const float4 x0 = ((const float4*)(in + base))[0];
    const float4 x1 = ((const float4*)(in + base))[1];
    v8h o;
    o[0] = (_Float16)x0.x; o[1] = (_Float16)x0.y;
    o[2] = (_Float16)x0.z; o[3] = (_Float16)x0.w;
    o[4] = (_Float16)x1.x; o[5] = (_Float16)x1.y;
    o[6] = (_Float16)x1.z; o[7] = (_Float16)x1.w;
    *(v8h*)(out + base) = o;
}

// ---------------------------------------------------------------------------
// Projection: Y = X @ W^T + bias, all-f16 operands, f32 WMMA accumulation.
// One wave -> 32x64 output tile (2 A fragments x 4 B fragments -> 8 WMMAs
// per K-step, 12 16B loads per K-step = 1.5 loads/WMMA; B fragments reused
// across both row groups, halving W traffic from L2 vs a 16-row tile).
// mode 0 -> [B,H,S,DH] (Q,K); mode 1 -> [B,H,DH,S] (V transposed).
// ---------------------------------------------------------------------------
__global__ void proj_kernel(const _Float16* __restrict__ X16, // [4096,1024]
                            const _Float16* __restrict__ W16, // [1024,1024]
                            const float* __restrict__ bias,
                            _Float16* __restrict__ out, int mode) {
    const int lane = threadIdx.x & 31;
    const int wave = threadIdx.x >> 5;
    const int job  = blockIdx.x * 8 + wave;       // 2048 wave-jobs
    const int NT4  = (D_MODEL / 16) / 4;          // 16 column groups
    const int tm   = job / NT4;                   // 0..127 (32-row tiles)
    const int tg   = job % NT4;
    const int row0 = tm * 32;
    const int col0 = tg * 64;
    const int lo   = lane & 15;
    const int hi   = lane >> 4;

    const _Float16* xr0 = X16 + (long)(row0 +      lo) * D_MODEL;
    const _Float16* xr1 = X16 + (long)(row0 + 16 + lo) * D_MODEL;
    const _Float16* wr0 = W16 + (long)(col0 +  0 + lo) * D_MODEL;
    const _Float16* wr1 = W16 + (long)(col0 + 16 + lo) * D_MODEL;
    const _Float16* wr2 = W16 + (long)(col0 + 32 + lo) * D_MODEL;
    const _Float16* wr3 = W16 + (long)(col0 + 48 + lo) * D_MODEL;

    v8f acc00 = {}, acc01 = {}, acc02 = {}, acc03 = {};
    v8f acc10 = {}, acc11 = {}, acc12 = {}, acc13 = {};

    for (int k0 = 0; k0 < D_MODEL; k0 += 32) {
        const int ka = k0 + hi * 16;  // A: contiguous 16-half row segment
        const int kb = k0 + hi * 16;  // B: contiguous 16-half column segment
        // Issue ALL loads of this K-step into distinct registers first.
        const v8h a0l = *(const v8h*)(xr0 + ka), a0h = *(const v8h*)(xr0 + ka + 8);
        const v8h a1l = *(const v8h*)(xr1 + ka), a1h = *(const v8h*)(xr1 + ka + 8);
        const v8h b0l = *(const v8h*)(wr0 + kb), b0h = *(const v8h*)(wr0 + kb + 8);
        const v8h b1l = *(const v8h*)(wr1 + kb), b1h = *(const v8h*)(wr1 + kb + 8);
        const v8h b2l = *(const v8h*)(wr2 + kb), b2h = *(const v8h*)(wr2 + kb + 8);
        const v8h b3l = *(const v8h*)(wr3 + kb), b3h = *(const v8h*)(wr3 + kb + 8);
        // WGP-scope prefetch of the next K-step (locality 3 -> all cache levels).
        __builtin_prefetch(xr0 + ka + 32, 0, 3);
        __builtin_prefetch(xr1 + ka + 32, 0, 3);
        __builtin_prefetch(wr0 + kb + 32, 0, 3);
        __builtin_prefetch(wr1 + kb + 32, 0, 3);
        __builtin_prefetch(wr2 + kb + 32, 0, 3);
        __builtin_prefetch(wr3 + kb + 32, 0, 3);

        const v16h a0 = combine(a0l, a0h);
        const v16h a1 = combine(a1l, a1h);
        const v16h b0 = combine(b0l, b0h);
        const v16h b1 = combine(b1l, b1h);
        const v16h b2 = combine(b2l, b2h);
        const v16h b3 = combine(b3l, b3h);
        acc00 = wmma16x16x32(a0, b0, acc00);
        acc10 = wmma16x16x32(a1, b0, acc10);
        acc01 = wmma16x16x32(a0, b1, acc01);
        acc11 = wmma16x16x32(a1, b1, acc11);
        acc02 = wmma16x16x32(a0, b2, acc02);
        acc12 = wmma16x16x32(a1, b2, acc12);
        acc03 = wmma16x16x32(a0, b3, acc03);
        acc13 = wmma16x16x32(a1, b3, acc13);
    }

#pragma unroll
    for (int rg = 0; rg < 2; ++rg) {
#pragma unroll
        for (int t = 0; t < 4; ++t) {
            const v8f acc =
                (rg == 0) ? ((t == 0) ? acc00 : (t == 1) ? acc01 : (t == 2) ? acc02 : acc03)
                          : ((t == 0) ? acc10 : (t == 1) ? acc11 : (t == 2) ? acc12 : acc13);
            const int ncol = col0 + t * 16 + lo;
            const int h  = ncol / D_HEAD;
            const int dh = ncol % D_HEAD;
            const float bb = bias[ncol];
#pragma unroll
            for (int v = 0; v < 8; ++v) {
                const int m = row0 + rg * 16 + v + hi * 8;
                const int b = m >> 11;           // / SEQ
                const int s = m & (SEQ - 1);
                const float val = acc[v] + bb;
                long idx;
                if (mode == 0) idx = ((long)(b * NUM_HEADS + h) * SEQ + s) * D_HEAD + dh;
                else           idx = ((long)(b * NUM_HEADS + h) * D_HEAD + dh) * SEQ + s;
                out[idx] = (_Float16)val;
            }
        }
    }
}

// ---------------------------------------------------------------------------
// Fused attention: one workgroup per (b, h, 16-query-row block).
//   Stage Q rows into LDS via global_load_async_to_lds (ASYNCcnt path).
//   Phase 1: scores 16x2048 via WMMA -> LDS (f32).
//   Phase 2: row softmax; float4 write of normalized attn (the 537MB stream,
//            written exactly once); f16 copy kept in LDS.
//   Phase 3: ctx = P @ V via WMMA, K split across waves, LDS reduction.
// ---------------------------------------------------------------------------
__global__ void attn_kernel(const _Float16* __restrict__ Qh,  // [B,H,S,DH]
                            const _Float16* __restrict__ Kh,  // [B,H,S,DH]
                            const _Float16* __restrict__ Vt,  // [B,H,DH,S]
                            float* __restrict__ ctx,          // [B,S,D]
                            float* __restrict__ attn)         // [H*B,S,S]
{
    extern __shared__ char lds[];
    float*    sc     = (float*)lds;                 // [16][2048]  131072 B
    _Float16* sp     = (_Float16*)(lds + 131072);   // [16][2048]   65536 B
    float*    red    = (float*)(lds + 196608);      // [16][16]
    float*    rowmax = (float*)(lds + 197632);      // [16]
    float*    rowsum = (float*)(lds + 197696);      // [16]
    float*    part   = (float*)(lds + 197760);      // [4][16][16]
    const _Float16* qlds = (const _Float16*)(lds + QLDS_OFF); // [16][64] 2048 B

    const int tid  = threadIdx.x;
    const int lane = tid & 31;
    const int wave = tid >> 5;
    const int lo   = lane & 15;
    const int hi   = lane >> 4;

    const int qblk = blockIdx.x;
    const int q0   = (qblk % (SEQ / 16)) * 16;
    const int bh   = qblk / (SEQ / 16);
    const int b    = bh / NUM_HEADS;
    const int h    = bh % NUM_HEADS;

    const _Float16* Kbase = Kh + (long)bh * SEQ * D_HEAD;
    const _Float16* Vbase = Vt + (long)bh * D_HEAD * SEQ;

    // ---- Stage the shared 16x64 Q tile (2KB) into LDS with async DMA.
    // All 8 waves consume the same A fragments -> fetch from HBM/L2 once.
    if (wave == 0) {
        const char* qg = (const char*)(Qh + (long)bh * SEQ * D_HEAD
                                          + (long)q0 * D_HEAD);
#pragma unroll
        for (int i = 0; i < 4; ++i) {
            const int loff = QLDS_OFF + lane * 16 + i * 512;
            const char* ga = qg + lane * 16 + i * 512;
            asm volatile("global_load_async_to_lds_b128 %0, %1, off"
                         :: "v"(loff), "v"(ga) : "memory");
        }
        asm volatile("s_wait_asynccnt 0x0" ::: "memory");
    }
    __syncthreads();

    // ---------------- Phase 1: scores ----------------
    const _Float16* qr = qlds + lo * D_HEAD;
    const v16h a0 = combine(*(const v8h*)(qr + hi * 16),
                            *(const v8h*)(qr + hi * 16 + 8));
    const v16h a1 = combine(*(const v8h*)(qr + 32 + hi * 16),
                            *(const v8h*)(qr + 32 + hi * 16 + 8));
    const float scale = 0.125f;  // 1/sqrt(D_HEAD)

    for (int t = 0; t < 16; ++t) {               // each wave covers 256 columns
        const int c0 = wave * 256 + t * 16;
        const _Float16* kr = Kbase + (long)(c0 + lo) * D_HEAD;
        const v8h b0l = *(const v8h*)(kr + hi * 16);
        const v8h b0h = *(const v8h*)(kr + hi * 16 + 8);
        const v8h b1l = *(const v8h*)(kr + 32 + hi * 16);
        const v8h b1h = *(const v8h*)(kr + 32 + hi * 16 + 8);
        __builtin_prefetch(kr + 16 * D_HEAD, 0, 3);   // next column tile -> WGP$
        v8f c = {};
        c = wmma16x16x32(a0, combine(b0l, b0h), c);
        c = wmma16x16x32(a1, combine(b1l, b1h), c);
#pragma unroll
        for (int v = 0; v < 8; ++v)
            sc[(v + hi * 8) * SEQ + c0 + lo] = c[v] * scale;
    }
    __syncthreads();

    // ---------------- Phase 2: softmax + single attn write ----------------
    {
        const int r   = tid >> 4;       // row 0..15
        const int sub = tid & 15;       // 16 threads per row
        float4* scr4 = (float4*)(sc + r * SEQ);   // 512 float4 per row

        float mx = -1e30f;
        for (int ci = 0; ci < SEQ / 64; ++ci) {
            const float4 v = scr4[ci * 16 + sub];
            mx = fmaxf(mx, fmaxf(fmaxf(v.x, v.y), fmaxf(v.z, v.w)));
        }
        red[r * 16 + sub] = mx;
        __syncthreads();
        if (sub == 0) {
            float m2 = red[r * 16];
#pragma unroll
            for (int i = 1; i < 16; ++i) m2 = fmaxf(m2, red[r * 16 + i]);
            rowmax[r] = m2;
        }
        __syncthreads();

        const float rm = rowmax[r];
        float s = 0.f;
        for (int ci = 0; ci < SEQ / 64; ++ci) {
            float4 v = scr4[ci * 16 + sub];
            v.x = __expf(v.x - rm); v.y = __expf(v.y - rm);
            v.z = __expf(v.z - rm); v.w = __expf(v.w - rm);
            s += (v.x + v.y) + (v.z + v.w);
            scr4[ci * 16 + sub] = v;
        }
        red[r * 16 + sub] = s;
        __syncthreads();
        if (sub == 0) {
            float t2 = 0.f;
#pragma unroll
            for (int i = 0; i < 16; ++i) t2 += red[r * 16 + i];
            rowsum[r] = t2;
        }
        __syncthreads();

        const float inv = 1.0f / rowsum[r];
        float4* arow4 = (float4*)(attn + ((long)(h * BATCH + b) * SEQ + (q0 + r)) * SEQ);
        _Float16* spr = sp + r * SEQ;
        for (int ci = 0; ci < SEQ / 64; ++ci) {   // coalesced float4 stream
            float4 v = scr4[ci * 16 + sub];
            v.x *= inv; v.y *= inv; v.z *= inv; v.w *= inv;
            arow4[ci * 16 + sub] = v;             // the one pass over 537MB
            v4h p;
            p[0] = (_Float16)v.x; p[1] = (_Float16)v.y;
            p[2] = (_Float16)v.z; p[3] = (_Float16)v.w;
            *(v4h*)(spr + (ci * 16 + sub) * 4) = p;
        }
    }
    __syncthreads();

    // ---------------- Phase 3: ctx = P @ V ----------------
    const int tn = wave & 3;                     // DH column tile
    const int kh = wave >> 2;                    // K half
    const int n0 = tn * 16;
    const int kbase = kh * (SEQ / 2);

    v8f cacc = {};
    for (int kk0 = kbase; kk0 < kbase + SEQ / 2; kk0 += 32) {
        const _Float16* ar = sp + lo * SEQ + kk0;
        const _Float16* vr = Vbase + (long)(n0 + lo) * SEQ + kk0;
        const v8h alo = *(const v8h*)(ar + hi * 16);
        const v8h ahi = *(const v8h*)(ar + hi * 16 + 8);
        const v8h blo = *(const v8h*)(vr + hi * 16);
        const v8h bhi = *(const v8h*)(vr + hi * 16 + 8);
        __builtin_prefetch(vr + 32, 0, 3);
        cacc = wmma16x16x32(combine(alo, ahi), combine(blo, bhi), cacc);
    }

    if (kh == 1) {
        float* pp = part + tn * 256;
#pragma unroll
        for (int v = 0; v < 8; ++v)
            pp[(v + hi * 8) * 16 + lo] = cacc[v];
    }
    __syncthreads();
    if (kh == 0) {
        float* pp = part + tn * 256;
#pragma unroll
        for (int v = 0; v < 8; ++v) {
            const int m = v + hi * 8;
            const float val = cacc[v] + pp[m * 16 + lo];
            ctx[((long)(b * SEQ + q0 + m) * D_MODEL) + h * D_HEAD + n0 + lo] = val;
        }
    }
}

// ---------------------------------------------------------------------------
extern "C" void kernel_launch(void* const* d_in, const int* in_sizes, int n_in,
                              void* d_out, int out_size, void* d_ws, size_t ws_size,
                              hipStream_t stream) {
    (void)in_sizes; (void)n_in; (void)out_size; (void)ws_size;

    const float* query = (const float*)d_in[0];
    const float* key   = (const float*)d_in[1];
    const float* value = (const float*)d_in[2];
    const float* Wq    = (const float*)d_in[3];
    const float* bq    = (const float*)d_in[4];
    const float* Wk    = (const float*)d_in[5];
    const float* bk    = (const float*)d_in[6];
    const float* Wv    = (const float*)d_in[7];
    const float* bv    = (const float*)d_in[8];

    float* ctx  = (float*)d_out;                       // [B,S,D]
    float* attn = ctx + (long)BS * D_MODEL;            // [H*B,S,S]

    const long NX = (long)BS * D_MODEL;                // 4M elems
    const long NW = (long)D_MODEL * D_MODEL;           // 1M elems
    _Float16* Xq16 = (_Float16*)d_ws;
    _Float16* Xk16 = Xq16 + NX;
    _Float16* Xv16 = Xk16 + NX;
    _Float16* Wq16 = Xv16 + NX;
    _Float16* Wk16 = Wq16 + NW;
    _Float16* Wv16 = Wk16 + NW;
    _Float16* Qh   = Wv16 + NW;                        // [B,H,S,DH]
    _Float16* Kh   = Qh + NX;                          // [B,H,S,DH]
    _Float16* Vt   = Kh + NX;                          // [B,H,DH,S]

    const dim3 blk(256);   // 8 waves (wave32)

    const int cvtX = (int)((NX / 8 + 255) / 256);      // 2048
    const int cvtW = (int)((NW / 8 + 255) / 256);      // 512
    cvt_kernel<<<cvtX, blk, 0, stream>>>(query, Xq16, NX / 8);
    cvt_kernel<<<cvtX, blk, 0, stream>>>(key,   Xk16, NX / 8);
    cvt_kernel<<<cvtX, blk, 0, stream>>>(value, Xv16, NX / 8);
    cvt_kernel<<<cvtW, blk, 0, stream>>>(Wq,    Wq16, NW / 8);
    cvt_kernel<<<cvtW, blk, 0, stream>>>(Wk,    Wk16, NW / 8);
    cvt_kernel<<<cvtW, blk, 0, stream>>>(Wv,    Wv16, NW / 8);

    const int projWGs = ((BS / 32) * (D_MODEL / 64)) / 8;  // 256
    proj_kernel<<<projWGs, blk, 0, stream>>>(Xq16, Wq16, bq, Qh, 0);
    proj_kernel<<<projWGs, blk, 0, stream>>>(Xk16, Wk16, bk, Kh, 0);
    proj_kernel<<<projWGs, blk, 0, stream>>>(Xv16, Wv16, bv, Vt, 1);

    const int attnWGs = BATCH * NUM_HEADS * (SEQ / 16);    // 4096
    attn_kernel<<<attnWGs, blk, LDS_BYTES, stream>>>(Qh, Kh, Vt, ctx, attn);
}
